// GPT_4690104287624
// MI455X (gfx1250) — matrix operand order repaired and checked by reading, not compiled, not run on hardware
//
#include <hip/hip_runtime.h>
#include <hip/hip_bf16.h>
#include <math.h>

// ---------------------------------------------------------------------------
// GPT-2 small forward on gfx1250 (MI455X).
// All GEMMs via v_wmma_f32_16x16x32_f16, fp32 accumulate.
// Model constants fixed by the reference: B=2 T=1024 C=768 H=12 D=64 L=12
// V=50257 (padded to 50304 for the head GEMM).
// Wave tile 32x64 (2x4 WMMA) -> 1.5 b128 loads per WMMA from L2-resident
// operands; block tile 128x128 (8 waves, 4x2).
// ---------------------------------------------------------------------------

typedef _Float16 half_t;
typedef __attribute__((ext_vector_type(16))) _Float16 v16h;
typedef __attribute__((ext_vector_type(8)))  _Float16 v8h;
typedef __attribute__((ext_vector_type(8)))  float    v8f;

#define LN_EPS 1e-3f

// ---------------------------------------------------------------------------
// Embedding: x[b,t,:] = wte[idx[b,t],:] + wpe[t,:]
// ---------------------------------------------------------------------------
__global__ __launch_bounds__(256) void embed_kernel(
    const int* __restrict__ idx, const float* __restrict__ wte,
    const float* __restrict__ wpe, float* __restrict__ x) {
  int bt = blockIdx.x;
  int t  = bt & 1023;
  int id = idx[bt];
  const float* wt = wte + (size_t)id * 768;
  const float* wp = wpe + (size_t)t * 768;
  float* xr = x + (size_t)bt * 768;
  for (int c = threadIdx.x; c < 768; c += 256) xr[c] = wt[c] + wp[c];
}

// ---------------------------------------------------------------------------
// LayerNorm (eps=1e-3) fp32 in -> f16 out (feeds WMMA A operand).
// ---------------------------------------------------------------------------
__global__ __launch_bounds__(256) void lnorm_f16_kernel(
    const float* __restrict__ x, const float* __restrict__ sc,
    const float* __restrict__ bi, half_t* __restrict__ out) {
  __shared__ float r1[256];
  __shared__ float r2[256];
  int row = blockIdx.x;
  int t = threadIdx.x;
  const float* xr = x + (size_t)row * 768;
  float v0 = xr[t], v1 = xr[t + 256], v2 = xr[t + 512];
  r1[t] = v0 + v1 + v2;
  r2[t] = v0 * v0 + v1 * v1 + v2 * v2;
  __syncthreads();
  for (int s = 128; s > 0; s >>= 1) {
    if (t < s) { r1[t] += r1[t + s]; r2[t] += r2[t + s]; }
    __syncthreads();
  }
  float mean = r1[0] * (1.0f / 768.0f);
  float var  = r2[0] * (1.0f / 768.0f) - mean * mean;
  float rs   = rsqrtf(var + LN_EPS);
  half_t* o = out + (size_t)row * 768;
  o[t]       = (half_t)((v0 - mean) * rs * sc[t]       + bi[t]);
  o[t + 256] = (half_t)((v1 - mean) * rs * sc[t + 256] + bi[t + 256]);
  o[t + 512] = (half_t)((v2 - mean) * rs * sc[t + 512] + bi[t + 512]);
}

// ---------------------------------------------------------------------------
// Weight convert + transpose: W fp32 [K,N] row-major -> WT f16 [Npad,K].
// Rows n in [N, Npad) zero-filled (head padding).  Batched over z (layers).
// ---------------------------------------------------------------------------
__global__ __launch_bounds__(256) void transpose_f32_f16_kernel(
    const float* __restrict__ W, half_t* __restrict__ WT,
    int K, int N, int Npad, size_t sIn, size_t sOut) {
  __shared__ float tile[32][33];
  const float* Wp = W  + (size_t)blockIdx.z * sIn;
  half_t*      Tp = WT + (size_t)blockIdx.z * sOut;
  int n0 = blockIdx.x * 32, k0 = blockIdx.y * 32;
  int tx = threadIdx.x, ty = threadIdx.y;
  for (int i = 0; i < 32; i += 8) {
    int k = k0 + ty + i, n = n0 + tx;
    tile[ty + i][tx] = (k < K && n < N) ? Wp[(size_t)k * N + n] : 0.0f;
  }
  __syncthreads();
  for (int i = 0; i < 32; i += 8) {
    int n = n0 + ty + i, k = k0 + tx;
    if (n < Npad && k < K) Tp[(size_t)n * K + k] = (half_t)tile[tx][ty + i];
  }
}

// ---------------------------------------------------------------------------
// Split QKV fp32 [B,T,3C] -> f16 Q,K [BH,T,D] and V-transposed [BH,D,T].
// ---------------------------------------------------------------------------
__global__ __launch_bounds__(256) void split_qkv_kernel(
    const float* __restrict__ qkv, half_t* __restrict__ q_h,
    half_t* __restrict__ k_h, half_t* __restrict__ vT_h) {
  int bt = blockIdx.x;
  int b = bt >> 10, t = bt & 1023;
  const float* row = qkv + (size_t)bt * 2304;
  for (int c = threadIdx.x; c < 2304; c += 256) {
    float v = row[c];
    int which = c / 768;
    int cc = c - which * 768;
    int h = cc >> 6, d = cc & 63;
    size_t bh = (size_t)(b * 12 + h);
    half_t hv = (half_t)v;
    if (which == 0)      q_h[(bh * 1024 + t) * 64 + d] = hv;
    else if (which == 1) k_h[(bh * 1024 + t) * 64 + d] = hv;
    else                 vT_h[(bh * 64 + d) * 1024 + t] = hv;
  }
}

// ---------------------------------------------------------------------------
// Causal softmax over scores fp32 [BH,T,T]: scale 1/8, mask k>q, -> f16 probs
// (zeros written for masked region so the P.V GEMM is correct).
// ---------------------------------------------------------------------------
__global__ __launch_bounds__(256) void softmax_kernel(
    const float* __restrict__ sc, half_t* __restrict__ p) {
  __shared__ float red[256];
  int r = blockIdx.x;
  int q = r & 1023;
  const float* srow = sc + (size_t)r * 1024;
  half_t* prow = p + (size_t)r * 1024;
  int t = threadIdx.x;
  float v[4];
  float mx = -3.4e38f;
  #pragma unroll
  for (int i = 0; i < 4; ++i) {
    int k = t + i * 256;
    float val = (k <= q) ? srow[k] * 0.125f : -3.4e38f;
    v[i] = val;
    mx = fmaxf(mx, val);
  }
  red[t] = mx; __syncthreads();
  for (int s = 128; s > 0; s >>= 1) {
    if (t < s) red[t] = fmaxf(red[t], red[t + s]);
    __syncthreads();
  }
  float rowMax = red[0];
  __syncthreads();
  float sum = 0.0f;
  #pragma unroll
  for (int i = 0; i < 4; ++i) {
    float e = __expf(v[i] - rowMax);   // masked lanes underflow to 0
    v[i] = e;
    sum += e;
  }
  red[t] = sum; __syncthreads();
  for (int s = 128; s > 0; s >>= 1) {
    if (t < s) red[t] += red[t + s];
    __syncthreads();
  }
  float inv = 1.0f / red[0];
  #pragma unroll
  for (int i = 0; i < 4; ++i) prow[t + i * 256] = (half_t)(v[i] * inv);
}

// ---------------------------------------------------------------------------
// WMMA GEMM:  Out[m,n] = sum_k A[m,k] * BT[n,k]  (+ bias / epilogue)
// A: f16 [M,K] row-major (lda).  BT: f16 [N,K] row-major (ldb) = B transposed.
// 256 threads = WR*WC = 8 waves; each wave computes (MT*16) x (NT*16) via
// MT x NT v_wmma_f32_16x16x32_f16 tiles, K-step 32.
// Block tile = (WR*MT*16) x (WC*NT*16).
// Batched over blockIdx.z with strides sA/sB/sO (elements).
// EPI: 0=store f32 (col<nReal bound), 1=bias+store f32, 2=bias+add f32,
//      3=bias+GELU(exact)->f16, 4=f16 store into y[B,T,H,D] (attn*V merge).
// flags: bit0 = causal wave skip (scores), bit1 = clamp K loop to causal
//        extent of this wave's rows (attn*V).
// ---------------------------------------------------------------------------
__device__ inline v16h make_frag(const half_t* p, int k) {
  v8h lo = *(const v8h*)(p + k);
  v8h hi = *(const v8h*)(p + k + 16);
  v16h r;
  #pragma unroll
  for (int j = 0; j < 8; ++j) { r[j] = lo[j]; r[j + 8] = hi[j]; }
  return r;
}

template <int WR, int WC, int MT, int NT, int EPI>
__global__ __launch_bounds__(256) void gemm_f16_wmma(
    const half_t* __restrict__ A, const half_t* __restrict__ B,
    const float* __restrict__ bias, float* __restrict__ outF,
    half_t* __restrict__ outH, int K, int lda, int ldb, int ldo, int nReal,
    size_t sA, size_t sB, size_t sO, int flags) {
  constexpr int WTM = MT * 16, WTN = NT * 16;
  constexpr int BM = WR * WTM, BN = WC * WTN;
  int lane = threadIdx.x & 31;
  int w = threadIdx.x >> 5;
  int mBase = blockIdx.x * BM + (w / WC) * WTM;
  int nBase = blockIdx.y * BN + (w % WC) * WTN;

  if ((flags & 1) && nBase >= mBase + WTM) return;  // fully-masked causal tile
  int kEnd = K;
  if (flags & 2) { int lim = mBase + WTM; kEnd = lim < K ? lim : K; }

  const half_t* Ap = A + (size_t)blockIdx.z * sA;
  const half_t* Bp = B + (size_t)blockIdx.z * sB;

  int mL  = lane & 15;
  int hiA = (lane & 16) ? 8 : 0;    // A frag: lanes 16-31 hold K+8 / K+24
  int hiB = (lane & 16) ? 8 : 0;    // B frag: lanes 16-31 hold K+8 / K+24
  const half_t* aBase[MT];
  const half_t* bBase[NT];
  #pragma unroll
  for (int mt = 0; mt < MT; ++mt)
    aBase[mt] = Ap + (size_t)(mBase + mt * 16 + mL) * lda + hiA;
  #pragma unroll
  for (int nt = 0; nt < NT; ++nt)
    bBase[nt] = Bp + (size_t)(nBase + nt * 16 + mL) * ldb + hiB;

  v8f acc[MT][NT] = {};
  for (int k = 0; k < kEnd; k += 32) {
    v16h a[MT], b[NT];
    #pragma unroll
    for (int mt = 0; mt < MT; ++mt) a[mt] = make_frag(aBase[mt], k);
    #pragma unroll
    for (int nt = 0; nt < NT; ++nt) b[nt] = make_frag(bBase[nt], k);
    #pragma unroll
    for (int mt = 0; mt < MT; ++mt)
      #pragma unroll
      for (int nt = 0; nt < NT; ++nt)
        acc[mt][nt] = __builtin_amdgcn_wmma_f32_16x16x32_f16(
            false, a[mt], false, b[nt], (short)0, acc[mt][nt], false, false);
  }

  int rowOff = (lane & 16) ? 8 : 0;
  size_t oOff = (EPI == 4)
      ? ((size_t)(blockIdx.z / 12) * (size_t)(1024 * 768) +
         (size_t)(blockIdx.z % 12) * 64)
      : (size_t)blockIdx.z * sO;

  #pragma unroll
  for (int mt = 0; mt < MT; ++mt) {
    #pragma unroll
    for (int nt = 0; nt < NT; ++nt) {
      int col  = nBase + nt * 16 + mL;
      int row0 = mBase + mt * 16 + rowOff;
      float bv = (EPI == 1 || EPI == 2 || EPI == 3) ? bias[col] : 0.0f;
      #pragma unroll
      for (int r = 0; r < 8; ++r) {
        float val = acc[mt][nt][r] + bv;
        size_t o = oOff + (size_t)(row0 + r) * ldo + col;
        if (EPI == 0) { if (col < nReal) outF[o] = val; }
        else if (EPI == 1) { outF[o] = val; }
        else if (EPI == 2) { outF[o] += val; }
        else if (EPI == 3) {
          float g = 0.5f * val * (1.0f + erff(val * 0.70710678118654752f));
          outH[o] = (half_t)g;
        } else { outH[o] = (half_t)val; }
      }
    }
  }
}

// ---------------------------------------------------------------------------
// Host launcher
// ---------------------------------------------------------------------------
extern "C" void kernel_launch(void* const* d_in, const int* in_sizes, int n_in,
                              void* d_out, int out_size, void* d_ws, size_t ws_size,
                              hipStream_t stream) {
  (void)in_sizes; (void)n_in; (void)out_size; (void)ws_size;

  const int*   idx    = (const int*)d_in[0];
  const float* wte    = (const float*)d_in[1];
  const float* wpe    = (const float*)d_in[2];
  const float* ln1_s  = (const float*)d_in[3];
  const float* ln1_b  = (const float*)d_in[4];
  const float* qkv_w  = (const float*)d_in[5];
  const float* qkv_b  = (const float*)d_in[6];
  const float* proj_w = (const float*)d_in[7];
  const float* proj_b = (const float*)d_in[8];
  const float* ln2_s  = (const float*)d_in[9];
  const float* ln2_b  = (const float*)d_in[10];
  const float* fc_w   = (const float*)d_in[11];
  const float* fc_b   = (const float*)d_in[12];
  const float* fc2_w  = (const float*)d_in[13];
  const float* fc2_b  = (const float*)d_in[14];
  const float* lnf_s  = (const float*)d_in[15];
  const float* lnf_b  = (const float*)d_in[16];
  const float* head_w = (const float*)d_in[17];
  float* out = (float*)d_out;

  // ---- workspace carve-out (~452 MB total) ----
  char* ws = (char*)d_ws;
  size_t off = 0;
  auto alloc = [&](size_t bytes) -> void* {
    void* p = ws + off;
    off += (bytes + 255) & ~(size_t)255;
    return p;
  };
  half_t* qkvT  = (half_t*)alloc((size_t)12 * 2304 * 768 * 2);   // [L][3C][C]
  half_t* projT = (half_t*)alloc((size_t)12 * 768  * 768 * 2);   // [L][C][C]
  half_t* fcT   = (half_t*)alloc((size_t)12 * 3072 * 768 * 2);   // [L][4C][C]
  half_t* fc2T  = (half_t*)alloc((size_t)12 * 768  * 3072 * 2);  // [L][C][4C]
  half_t* headT = (half_t*)alloc((size_t)50304 * 768 * 2);       // [Vpad][C]
  half_t* ln_h  = (half_t*)alloc((size_t)2048 * 768 * 2);
  half_t* q_h   = (half_t*)alloc((size_t)24 * 1024 * 64 * 2);
  half_t* k_h   = (half_t*)alloc((size_t)24 * 1024 * 64 * 2);
  half_t* vT_h  = (half_t*)alloc((size_t)24 * 64 * 1024 * 2);
  half_t* p_h   = (half_t*)alloc((size_t)24 * 1024 * 1024 * 2);
  half_t* y_h   = (half_t*)alloc((size_t)2048 * 768 * 2);
  half_t* h_h   = (half_t*)alloc((size_t)2048 * 3072 * 2);
  float*  x     = (float*)alloc((size_t)2048 * 768 * 4);
  float*  qkvf  = (float*)alloc((size_t)2048 * 2304 * 4);
  float*  scores= (float*)alloc((size_t)24 * 1024 * 1024 * 4);

  dim3 tb(32, 8);
  // ---- weight convert+transpose (f32 [K,N] -> f16 [Npad,K]) ----
  transpose_f32_f16_kernel<<<dim3(72, 24, 12), tb, 0, stream>>>(
      qkv_w, qkvT, 768, 2304, 2304, (size_t)768 * 2304, (size_t)2304 * 768);
  transpose_f32_f16_kernel<<<dim3(24, 24, 12), tb, 0, stream>>>(
      proj_w, projT, 768, 768, 768, (size_t)768 * 768, (size_t)768 * 768);
  transpose_f32_f16_kernel<<<dim3(96, 24, 12), tb, 0, stream>>>(
      fc_w, fcT, 768, 3072, 3072, (size_t)768 * 3072, (size_t)3072 * 768);
  transpose_f32_f16_kernel<<<dim3(24, 96, 12), tb, 0, stream>>>(
      fc2_w, fc2T, 3072, 768, 768, (size_t)3072 * 768, (size_t)768 * 3072);
  transpose_f32_f16_kernel<<<dim3(1572, 24, 1), tb, 0, stream>>>(
      head_w, headT, 768, 50257, 50304, 0, 0);

  // ---- embedding ----
  embed_kernel<<<2048, 256, 0, stream>>>(idx, wte, wpe, x);

  for (int l = 0; l < 12; ++l) {
    // LN1 -> f16
    lnorm_f16_kernel<<<2048, 256, 0, stream>>>(x, ln1_s + l * 768, ln1_b + l * 768, ln_h);
    // QKV: [2048x768] x [768x2304] + bias -> f32   (block 128x128)
    gemm_f16_wmma<4, 2, 2, 4, 1><<<dim3(16, 18, 1), 256, 0, stream>>>(
        ln_h, qkvT + (size_t)l * 2304 * 768, qkv_b + l * 2304,
        qkvf, nullptr, 768, 768, 768, 2304, 2304, 0, 0, 0, 0);
    // split into Q,K [BH,T,D] and V^T [BH,D,T] f16
    split_qkv_kernel<<<2048, 256, 0, stream>>>(qkvf, q_h, k_h, vT_h);
    // scores = Q.K^T (batched over BH), causal wave skip  (block 128x128)
    gemm_f16_wmma<4, 2, 2, 4, 0><<<dim3(8, 8, 24), 256, 0, stream>>>(
        q_h, k_h, nullptr, scores, nullptr, 64, 64, 64, 1024, 1024,
        (size_t)1024 * 64, (size_t)1024 * 64, (size_t)1024 * 1024, 1);
    // scale + causal softmax -> f16 probs (zeros above diagonal)
    softmax_kernel<<<24576, 256, 0, stream>>>(scores, p_h);
    // Y = P.V (batched), K clamped to causal extent, scatter into [B,T,C]
    gemm_f16_wmma<4, 2, 2, 2, 4><<<dim3(8, 1, 24), 256, 0, stream>>>(
        p_h, vT_h, nullptr, nullptr, y_h, 1024, 1024, 1024, 768, 64,
        (size_t)1024 * 1024, (size_t)64 * 1024, 0, 2);
    // proj + residual add into x  (block 128x128)
    gemm_f16_wmma<4, 2, 2, 4, 2><<<dim3(16, 6, 1), 256, 0, stream>>>(
        y_h, projT + (size_t)l * 768 * 768, proj_b + l * 768,
        x, nullptr, 768, 768, 768, 768, 768, 0, 0, 0, 0);
    // LN2 -> f16
    lnorm_f16_kernel<<<2048, 256, 0, stream>>>(x, ln2_s + l * 768, ln2_b + l * 768, ln_h);
    // FC + bias + exact GELU -> f16  (block 128x128)
    gemm_f16_wmma<4, 2, 2, 4, 3><<<dim3(16, 24, 1), 256, 0, stream>>>(
        ln_h, fcT + (size_t)l * 3072 * 768, fc_b + l * 3072,
        nullptr, h_h, 768, 768, 768, 3072, 3072, 0, 0, 0, 0);
    // FC2 + residual add into x  (block 128x128)
    gemm_f16_wmma<4, 2, 2, 4, 2><<<dim3(16, 6, 1), 256, 0, stream>>>(
        h_h, fc2T + (size_t)l * 768 * 3072, fc2_b + l * 768,
        x, nullptr, 3072, 3072, 3072, 768, 768, 0, 0, 0, 0);
  }

  // final LN + LM head (N padded to 50304; stores bounded at 50257)
  lnorm_f16_kernel<<<2048, 256, 0, stream>>>(x, lnf_s, lnf_b, ln_h);
  gemm_f16_wmma<4, 2, 2, 4, 0><<<dim3(16, 393, 1), 256, 0, stream>>>(
      ln_h, headT, nullptr, out, nullptr, 768, 768, 768, 50257, 50257,
      0, 0, 0, 0);
}
